// LocalSparseAttention_69810398429691
// MI455X (gfx1250) — matrix operand order repaired and checked by reading, not compiled, not run on hardware
//
#include <hip/hip_runtime.h>
#include <hip/hip_bf16.h>

#define S_LEN    4096
#define D_DIM    512
#define HALF_WIN 128

typedef __attribute__((ext_vector_type(16))) __bf16 v16bf;
typedef __attribute__((ext_vector_type(8)))  __bf16 v8bf;
typedef __attribute__((ext_vector_type(8)))  float  v8f;
typedef __attribute__((ext_vector_type(4)))  int    v4i_;

// ---- CDNA5 async global->LDS copy (ASYNCcnt path), with safe fallback ----
#if defined(__has_builtin)
#  if __has_builtin(__builtin_amdgcn_global_load_async_to_lds_b128)
#    define LSA_HAVE_ASYNC 1
#  endif
#endif
#ifndef LSA_HAVE_ASYNC
#  define LSA_HAVE_ASYNC 0
#endif

#if LSA_HAVE_ASYNC
typedef __attribute__((address_space(1))) v4i_ gv4i;   // global int4
typedef __attribute__((address_space(3))) v4i_ lv4i;   // LDS int4
#endif

static __device__ __forceinline__ void lsa_cp16(const __bf16* g, __bf16* l) {
#if LSA_HAVE_ASYNC
  __builtin_amdgcn_global_load_async_to_lds_b128((gv4i*)g, (lv4i*)l, 0, 0);
#else
  *(v8bf*)l = *(const v8bf*)g;
#endif
}

static __device__ __forceinline__ void lsa_async_wait() {
#if LSA_HAVE_ASYNC
#  if defined(__has_builtin) && __has_builtin(__builtin_amdgcn_s_wait_asynccnt)
  __builtin_amdgcn_s_wait_asynccnt(0);
#  else
  asm volatile("s_wait_asynccnt 0x0" ::: "memory");
#  endif
#endif
}

// Build a 16-element bf16 WMMA fragment from two 16-byte halves.
static __device__ __forceinline__ v16bf ld_frag16(const __bf16* p0, const __bf16* p1) {
  union { v16bf v; v8bf h[2]; } u;
  u.h[0] = *(const v8bf*)p0;
  u.h[1] = *(const v8bf*)p1;
  return u.v;
}

// ---------------- Phase 0: conversions ----------------

__global__ void lsa_cvt_x(const float* __restrict__ x, __bf16* __restrict__ xb) {
  int i = (blockIdx.x * blockDim.x + threadIdx.x) * 4;
  float4 f = *(const float4*)(x + i);
  xb[i + 0] = (__bf16)f.x;
  xb[i + 1] = (__bf16)f.y;
  xb[i + 2] = (__bf16)f.z;
  xb[i + 3] = (__bf16)f.w;
}

// wt[z][n][k] = W_z[k][n]  (transposed so B-fragment loads are contiguous)
__global__ void lsa_cvt_wT(const float* __restrict__ Wq, const float* __restrict__ Wk,
                           const float* __restrict__ Wv, __bf16* __restrict__ wt) {
  int i = blockIdx.x * blockDim.x + threadIdx.x;   // 0 .. 3*512*512-1
  int z = i >> 18;
  int r = i & 262143;
  int n = r >> 9;
  int k = r & 511;
  const float* W = (z == 0) ? Wq : ((z == 1) ? Wk : Wv);
  wt[i] = (__bf16)W[k * D_DIM + n];
}

// ---------------- Phase 1: Q/K/V projections ----------------
// One wave per 16x16 output tile; K = 512 -> 16 wmma_f32_16x16x32_bf16 steps.

__global__ void lsa_gemm_qkv(const __bf16* __restrict__ xb, const __bf16* __restrict__ wt,
                             const float* __restrict__ bq, const float* __restrict__ bk,
                             const float* __restrict__ bv,
                             __bf16* __restrict__ qb, __bf16* __restrict__ kb,
                             __bf16* __restrict__ vb) {
  const int z = blockIdx.z;
  const __bf16* wz   = wt + (size_t)z * D_DIM * D_DIM;
  const float*  bias = (z == 0) ? bq : ((z == 1) ? bk : bv);
  __bf16*       outp = (z == 0) ? qb : ((z == 1) ? kb : vb);

  const int lane = threadIdx.x & 31;
  const int wave = threadIdx.x >> 5;
  const int tile = blockIdx.x * 8 + wave;   // 32768 tiles = 1024 mtiles * 32 ntiles
  const int mt = tile >> 5;
  const int nt = tile & 31;

  const int lane15 = lane & 15;
  const int offA = (lane < 16) ? 0 : 16;    // A: k in {0..15} / {16..31}
  const int offB = (lane < 16) ? 0 : 16;    // B: k in {0..15} / {16..31}
  const __bf16* arow = xb + (size_t)(mt * 16 + lane15) * D_DIM;
  const __bf16* brow = wz + (size_t)(nt * 16 + lane15) * D_DIM;

  v8f c = {};
#pragma unroll
  for (int k = 0; k < 16; ++k) {
    v16bf a = ld_frag16(arow + k * 32 + offA, arow + k * 32 + offA + 8);
    v16bf b = ld_frag16(brow + k * 32 + offB, brow + k * 32 + offB + 8);
    c = __builtin_amdgcn_wmma_f32_16x16x32_bf16(false, a, false, b, (short)0, c, false, false);
  }

  const float bval = bias[nt * 16 + lane15];
  const int rowoff = (lane >> 4) << 3;      // C: vgpr r -> row r + 8*(lane/16)
#pragma unroll
  for (int r = 0; r < 8; ++r) {
    outp[(size_t)(mt * 16 + rowoff + r) * D_DIM + nt * 16 + lane15] = (__bf16)(c[r] + bval);
  }
}

// ---------------- Phase 2: banded flash attention ----------------
// Block = 256 threads (8 waves) per (batch, 32 queries).
// Wave (qi,dj): qi in {0,1} -> 16 query rows; dj in {0..3} -> 128 output cols.

#define QS_STR 520   // padded LDS strides (bank-conflict avoidance)
#define VT_STR 40
#define PS_STR 40

__global__ void lsa_attn(const __bf16* __restrict__ qb, const __bf16* __restrict__ kb,
                         const __bf16* __restrict__ vb, float* __restrict__ out) {
  const int b  = blockIdx.y;
  const int q0 = blockIdx.x * 32;
  const int tid  = threadIdx.x;
  const int lane = tid & 31;
  const int wave = tid >> 5;
  const int qi = wave & 1;
  const int dj = wave >> 1;

  __shared__ __bf16 Qs[32 * QS_STR];
  __shared__ __bf16 Ks[32 * QS_STR];
  __shared__ __bf16 Vts[D_DIM * VT_STR];     // transposed: Vts[d][t]
  __shared__ __bf16 Ps[8 * 16 * PS_STR];     // per-wave P scratch

  const size_t base = (size_t)b * S_LEN * D_DIM;

  // Stage Q tile (32 x 512 bf16): async global->LDS copies (8 x b128 per thread)
#pragma unroll
  for (int i = 0; i < 8; ++i) {
    int idx = tid + i * 256;
    int r   = idx >> 6;
    int c8  = (idx & 63) << 3;
    lsa_cp16(&qb[base + (size_t)(q0 + r) * D_DIM + c8], &Qs[r * QS_STR + c8]);
  }

  const float scale = 0.04419417382415922f;  // 1/sqrt(512)
  float m_prev[8], lsum[8], fac[8];
  v8f o[8];
#pragma unroll
  for (int r = 0; r < 8; ++r) { m_prev[r] = -1e30f; lsum[r] = 0.0f; v8f zz = {}; o[r] = zz; }

  const int lane15 = lane & 15;
  const int rowoff = (lane >> 4) << 3;
  const int offA = (lane < 16) ? 0 : 16;
  const int offB = (lane < 16) ? 0 : 16;
  const int qg = q0 + qi * 16;
  __bf16* Psw = Ps + wave * 16 * PS_STR;

  int tstart = q0 - HALF_WIN; if (tstart < 0) tstart = 0;
  int tend   = q0 + 32 + HALF_WIN; if (tend > S_LEN) tend = S_LEN;
  // chunks are always full 32 keys (q0 and HALF_WIN are multiples of 32)

  for (int t0 = tstart; t0 < tend; t0 += 32) {
    __syncthreads();   // previous chunk fully consumed (also orders Q staging issue)
    // Stage K (async, row-major) and V (VGPR round-trip, transposed) chunk
#pragma unroll
    for (int i = 0; i < 8; ++i) {
      int idx = tid + i * 256;
      int r   = idx >> 6;
      int c8  = (idx & 63) << 3;
      lsa_cp16(&kb[base + (size_t)(t0 + r) * D_DIM + c8], &Ks[r * QS_STR + c8]);
      v8bf v = *(const v8bf*)&vb[base + (size_t)(t0 + r) * D_DIM + c8];
#pragma unroll
      for (int e = 0; e < 8; ++e) Vts[(c8 + e) * VT_STR + r] = v[e];
    }
    lsa_async_wait();  // wave's async Q/K copies landed in LDS
    __syncthreads();

    // ---- Scores: Q(16x512) @ K^T(512x32) -> two 16x16 tiles ----
    v8f c0 = {}, c1 = {};
    const __bf16* qrow = &Qs[(qi * 16 + lane15) * QS_STR];
#pragma unroll
    for (int kd = 0; kd < 16; ++kd) {
      v16bf a = ld_frag16(qrow + kd * 32 + offA, qrow + kd * 32 + offA + 8);
      const __bf16* k0 = &Ks[lane15 * QS_STR + kd * 32 + offB];
      const __bf16* k1 = &Ks[(16 + lane15) * QS_STR + kd * 32 + offB];
      v16bf b0 = ld_frag16(k0, k0 + 8);
      v16bf b1 = ld_frag16(k1, k1 + 8);
      c0 = __builtin_amdgcn_wmma_f32_16x16x32_bf16(false, a, false, b0, (short)0, c0, false, false);
      c1 = __builtin_amdgcn_wmma_f32_16x16x32_bf16(false, a, false, b1, (short)0, c1, false, false);
    }

    // ---- Online softmax (per row; 16-lane shuffle reductions) ----
#pragma unroll
    for (int r = 0; r < 8; ++r) {
      const int q  = qg + rowoff + r;
      int d0 = q - (t0 + lane15);      if (d0 < 0) d0 = -d0;
      int d1 = q - (t0 + 16 + lane15); if (d1 < 0) d1 = -d1;
      float s0 = (d0 <= HALF_WIN) ? c0[r] * scale : -1e30f;
      float s1 = (d1 <= HALF_WIN) ? c1[r] * scale : -1e30f;
      float mx = fmaxf(s0, s1);
      mx = fmaxf(mx, __shfl_xor(mx, 1));
      mx = fmaxf(mx, __shfl_xor(mx, 2));
      mx = fmaxf(mx, __shfl_xor(mx, 4));
      mx = fmaxf(mx, __shfl_xor(mx, 8));
      float m_new = fmaxf(m_prev[r], mx);
      fac[r] = __expf(m_prev[r] - m_new);
      float p0 = __expf(s0 - m_new);
      float p1 = __expf(s1 - m_new);
      float ps = p0 + p1;
      ps += __shfl_xor(ps, 1);
      ps += __shfl_xor(ps, 2);
      ps += __shfl_xor(ps, 4);
      ps += __shfl_xor(ps, 8);
      lsum[r]   = lsum[r] * fac[r] + ps;
      m_prev[r] = m_new;
      // restripe P into A-fragment layout via per-wave LDS scratch
      Psw[(rowoff + r) * PS_STR + lane15]      = (__bf16)p0;
      Psw[(rowoff + r) * PS_STR + 16 + lane15] = (__bf16)p1;
    }

    // rescale running output
#pragma unroll
    for (int j = 0; j < 8; ++j)
#pragma unroll
      for (int r = 0; r < 8; ++r) o[j][r] *= fac[r];

    // ---- PV: P(16x32) @ V(32x128 chunk) ----
    const __bf16* prow = Psw + lane15 * PS_STR;
    v16bf pa = ld_frag16(prow + offA, prow + offA + 8);
#pragma unroll
    for (int j = 0; j < 8; ++j) {
      const __bf16* vcol = &Vts[(dj * 128 + j * 16 + lane15) * VT_STR + offB];
      v16bf vf = ld_frag16(vcol, vcol + 8);
      o[j] = __builtin_amdgcn_wmma_f32_16x16x32_bf16(false, pa, false, vf, (short)0, o[j], false, false);
    }
  }

  // ---- Normalize and store f32 output ----
#pragma unroll
  for (int j = 0; j < 8; ++j) {
    const int col = dj * 128 + j * 16 + lane15;
#pragma unroll
    for (int r = 0; r < 8; ++r) {
      out[base + (size_t)(q0 + qi * 16 + rowoff + r) * D_DIM + col] = o[j][r] / lsum[r];
    }
  }
}

// ---------------- launch ----------------

extern "C" void kernel_launch(void* const* d_in, const int* in_sizes, int n_in,
                              void* d_out, int out_size, void* d_ws, size_t ws_size,
                              hipStream_t stream) {
  const float* x  = (const float*)d_in[0];
  const float* Wq = (const float*)d_in[1];
  const float* bq = (const float*)d_in[2];
  const float* Wk = (const float*)d_in[3];
  const float* bk = (const float*)d_in[4];
  const float* Wv = (const float*)d_in[5];
  const float* bv = (const float*)d_in[6];
  float* out = (float*)d_out;

  char* ws = (char*)d_ws;
  const size_t XB_BYTES = (size_t)4 * S_LEN * D_DIM * 2;      // 16.78 MB
  const size_t WT_BYTES = (size_t)3 * D_DIM * D_DIM * 2;      // 1.57 MB
  __bf16* xb = (__bf16*)ws;
  __bf16* wt = (__bf16*)(ws + XB_BYTES);
  __bf16* qb = (__bf16*)(ws + XB_BYTES + WT_BYTES);
  __bf16* kb = qb + (size_t)4 * S_LEN * D_DIM;
  __bf16* vb = kb + (size_t)4 * S_LEN * D_DIM;

  // Phase 0: conversions
  lsa_cvt_x<<<dim3(8192), dim3(256), 0, stream>>>(x, xb);            // 4*4096*512 / (256*4)
  lsa_cvt_wT<<<dim3(3072), dim3(256), 0, stream>>>(Wq, Wk, Wv, wt);  // 3*512*512 / 256

  // Phase 1: Q/K/V = x@W + b   (1024*32 tiles, 8 waves/block, z = matrix)
  lsa_gemm_qkv<<<dim3(4096, 1, 3), dim3(256), 0, stream>>>(xb, wt, bq, bk, bv, qb, kb, vb);

  // Phase 2: banded attention  (S/32 query blocks x B batches)
  lsa_attn<<<dim3(S_LEN / 32, 4), dim3(256), 0, stream>>>(qb, kb, vb, out);
}
